// minGRU_83434034692858
// MI455X (gfx1250) — matrix operand order repaired and checked by reading, not compiled, not run on hardware
//
#include <hip/hip_runtime.h>
#include <hip/hip_bf16.h>
#include <stdint.h>

// Problem dims (compile-time, from the reference)
#define DIMX      1024
#define DIM_INNER 2048
#define NHG       (2 * DIM_INNER)   // 4096
#define BATCH     4
#define SEQ       4096
#define M_TOT     (BATCH * SEQ)     // 16384
#define NCH       (BATCH * DIM_INNER)   // 8192 scan channels
#define NCHUNK    16
#define CLEN      (SEQ / NCHUNK)        // 256

typedef __bf16 bf16_t;
typedef __attribute__((ext_vector_type(16))) __bf16 v16bf;
typedef __attribute__((ext_vector_type(8)))  float  v8f;

struct FragRaw { uint4 lo; uint4 hi; };   // 32 bytes == v16bf

// ---- fp32 -> bf16 (round-to-nearest-even) --------------------------------
static __device__ __forceinline__ uint16_t f2bf(float x) {
  union { float f; uint32_t u; } v; v.f = x;
  uint32_t u = v.u;
  u += 0x7FFFu + ((u >> 16) & 1u);
  return (uint16_t)(u >> 16);
}

// ---- WMMA fragment loaders (bf16, K=32) ----------------------------------
// A: 16x32 row-major tile. Lanes 0-15: M=lane, K=0..15;
//    lanes 16-31: M=lane-16, K=16..31.
static __device__ __forceinline__ v16bf load_frag_a(const uint16_t* __restrict__ base,
                                                    int lda, int m0, int k0, int lane) {
  const int r  = lane & 15;
  const int kh = lane >> 4;            // 0 or 1
  const uint16_t* p = base + (size_t)(m0 + r) * lda + (k0 + kh * 16);
  FragRaw f;
  f.lo = *reinterpret_cast<const uint4*>(p);        // K = kh*16 .. +7
  f.hi = *reinterpret_cast<const uint4*>(p + 8);    // K = kh*16+8 .. +15
  return __builtin_bit_cast(v16bf, f);
}
// B: 32x16 tile fed from a PRE-TRANSPOSED weight (rows = N, cols = K).
// Lanes 0-15: N=lane, K=0..15; lanes 16-31: N=lane-16, K=16..31.
static __device__ __forceinline__ v16bf load_frag_b(const uint16_t* __restrict__ baseT,
                                                    int ldb, int n0, int k0, int lane) {
  const int n  = lane & 15;
  const int kh = lane >> 4;
  const uint16_t* p = baseT + (size_t)(n0 + n) * ldb + (k0 + kh * 16);
  FragRaw f;
  f.lo = *reinterpret_cast<const uint4*>(p);        // K = kh*16 .. +7
  f.hi = *reinterpret_cast<const uint4*>(p + 8);    // K = kh*16+8 .. +15
  return __builtin_bit_cast(v16bf, f);
}

// ---- Kernel 1: x fp32 -> bf16 --------------------------------------------
__global__ __launch_bounds__(256) void mg_convert_x(const float* __restrict__ in,
                                                    uint16_t* __restrict__ out) {
  size_t i = (size_t)blockIdx.x * blockDim.x + threadIdx.x;  // one float4 per thread
  const float4 v = reinterpret_cast<const float4*>(in)[i];
  ushort4 o;
  o.x = f2bf(v.x); o.y = f2bf(v.y); o.z = f2bf(v.z); o.w = f2bf(v.w);
  reinterpret_cast<ushort4*>(out)[i] = o;
}

// ---- Kernel 2: transpose + convert: in fp32 [R x C] -> out bf16 [C x R] --
__global__ __launch_bounds__(256) void mg_transpose_bf16(const float* __restrict__ in,
                                                         uint16_t* __restrict__ out,
                                                         int R, int C) {
  __shared__ float tile[32][33];
  const int x = blockIdx.x * 32 + threadIdx.x;   // input col
  const int y = blockIdx.y * 32 + threadIdx.y;   // input row
  #pragma unroll
  for (int j = 0; j < 32; j += 8)
    tile[threadIdx.y + j][threadIdx.x] = in[(size_t)(y + j) * C + x];
  __syncthreads();
  const int x2 = blockIdx.y * 32 + threadIdx.x;  // output col (= input row)
  const int y2 = blockIdx.x * 32 + threadIdx.y;  // output row (= input col)
  #pragma unroll
  for (int j = 0; j < 32; j += 8)
    out[(size_t)(y2 + j) * R + x2] = f2bf(tile[threadIdx.x][threadIdx.y + j]);
}

// ---- Kernel 3: fused GEMM1 (hidden+gate share A) + activation epilogue ---
__global__ __launch_bounds__(256) void mg_gemm1_act(const uint16_t* __restrict__ xb,    // [M_TOT, DIMX]
                                                    const uint16_t* __restrict__ whgT,  // [NHG,  DIMX]
                                                    float* __restrict__ logc,           // [M_TOT, DIM_INNER]
                                                    float* __restrict__ logv) {
  const int lane = threadIdx.x & 31;
  const int wid  = blockIdx.x * (blockDim.x >> 5) + (threadIdx.x >> 5);
  const int nTiles = DIM_INNER / 32;            // 64
  const int m0 = (wid / nTiles) * 64;
  const int n0 = (wid % nTiles) * 32;

  const v8f vzero = {0.f, 0.f, 0.f, 0.f, 0.f, 0.f, 0.f, 0.f};
  v8f acch[4][2], accg[4][2];
  #pragma unroll
  for (int mi = 0; mi < 4; ++mi)
    #pragma unroll
    for (int ni = 0; ni < 2; ++ni) { acch[mi][ni] = vzero; accg[mi][ni] = vzero; }

  for (int k0 = 0; k0 < DIMX; k0 += 32) {
    __builtin_prefetch(xb + (size_t)(m0 + (lane & 15)) * DIMX + k0 + 32, 0, 0);
    v16bf a[4];
    #pragma unroll
    for (int mi = 0; mi < 4; ++mi)
      a[mi] = load_frag_a(xb, DIMX, m0 + 16 * mi, k0, lane);
    #pragma unroll
    for (int ni = 0; ni < 2; ++ni) {
      v16bf bh = load_frag_b(whgT, DIMX, n0 + 16 * ni, k0, lane);
      v16bf bg = load_frag_b(whgT, DIMX, DIM_INNER + n0 + 16 * ni, k0, lane);
      #pragma unroll
      for (int mi = 0; mi < 4; ++mi) {
        acch[mi][ni] = __builtin_amdgcn_wmma_f32_16x16x32_bf16(
            false, a[mi], false, bh, (short)0, acch[mi][ni], false, false);
        accg[mi][ni] = __builtin_amdgcn_wmma_f32_16x16x32_bf16(
            false, a[mi], false, bg, (short)0, accg[mi][ni], false, false);
      }
    }
  }

  const int cn = lane & 15;
  const int cm = (lane >> 4) * 8;
  #pragma unroll
  for (int mi = 0; mi < 4; ++mi)
    #pragma unroll
    for (int ni = 0; ni < 2; ++ni)
      #pragma unroll
      for (int i = 0; i < 8; ++i) {
        const int row = m0 + 16 * mi + cm + i;
        const int col = n0 + 16 * ni + cn;
        const float hv = acch[mi][ni][i];
        const float gv = accg[mi][ni][i];
        const float sp = fmaxf(gv, 0.f) + log1pf(__expf(-fabsf(gv)));  // softplus(g)
        const size_t idx = (size_t)row * DIM_INNER + col;
        logc[idx] = -sp;                                               // log(1-sigmoid(g))
        logv[idx] = __logf(fmaxf(hv, 0.f) + 0.5f) + (gv - sp);         // log(relu(h)+.5)+logsig(g)
      }
}

// ---- Chunked scan, pass A: per-chunk summaries ---------------------------
// Thread g -> chunk j = g / NCH, channel c = g % NCH (coalesced over channels).
__global__ __launch_bounds__(256) void mg_scan_part(const float* __restrict__ logc,
                                                    const float* __restrict__ logv,
                                                    float* __restrict__ chunkA,   // [NCHUNK][NCH]
                                                    float* __restrict__ chunkL) { // [NCHUNK][NCH]
  const int g = blockIdx.x * blockDim.x + threadIdx.x;
  const int j = g / NCH;
  const int c = g % NCH;
  const int b  = c / DIM_INNER;
  const int ch = c % DIM_INNER;
  size_t idx = ((size_t)b * SEQ + (size_t)j * CLEN) * DIM_INNER + ch;
  float a = 0.f, m = -__builtin_inff(), s = 0.f;
  for (int t = 0; t < CLEN; ++t) {
    const float lc = logc[idx];
    const float lv = logv[idx];
    a += lc;
    const float u = lv - a;                 // chunk-local domain
    if (u <= m) s += __expf(u - m);
    else { s = s * __expf(m - u) + 1.f; m = u; }
    idx += DIM_INNER;
  }
  chunkA[(size_t)j * NCH + c] = a;          // sum of log_coeffs over chunk
  chunkL[(size_t)j * NCH + c] = m + __logf(s);  // chunk-local logsumexp
}

// ---- Chunked scan, pass B: 16-step carry scan per channel ----------------
__global__ __launch_bounds__(256) void mg_scan_carry(const float* __restrict__ chunkA,
                                                     const float* __restrict__ chunkL,
                                                     float* __restrict__ carryA,   // a_star prefix before chunk
                                                     float* __restrict__ carryG) { // global LSE before chunk
  const int c = blockIdx.x * blockDim.x + threadIdx.x;   // 0..NCH-1
  float Ap = 0.f;
  float G  = -__builtin_inff();
  #pragma unroll
  for (int j = 0; j < NCHUNK; ++j) {
    const size_t o = (size_t)j * NCH + c;
    carryA[o] = Ap;
    carryG[o] = G;
    const float L = chunkL[o] - Ap;          // chunk LSE re-based to global domain
    if (L <= G) G += log1pf(__expf(L - G));
    else        G  = L + log1pf(__expf(G - L));  // handles G = -inf
    Ap += chunkA[o];
  }
}

// ---- Chunked scan, pass C: apply carries, emit h (bf16) ------------------
__global__ __launch_bounds__(256) void mg_scan_apply(const float* __restrict__ logc,
                                                     const float* __restrict__ logv,
                                                     const float* __restrict__ carryA,
                                                     const float* __restrict__ carryG,
                                                     uint16_t* __restrict__ hb) {
  const int g = blockIdx.x * blockDim.x + threadIdx.x;
  const int j = g / NCH;
  const int c = g % NCH;
  const int b  = c / DIM_INNER;
  const int ch = c % DIM_INNER;
  const size_t o = (size_t)j * NCH + c;
  float a = carryA[o];                       // global a_star entering chunk
  const float G = carryG[o];
  float m = G;
  float s = (G == -__builtin_inff()) ? 0.f : 1.f;
  size_t idx = ((size_t)b * SEQ + (size_t)j * CLEN) * DIM_INNER + ch;
  for (int t = 0; t < CLEN; ++t) {
    const float lc = logc[idx];
    const float lv = logv[idx];
    a += lc;
    const float u = lv - a;                  // global domain
    if (u <= m) s += __expf(u - m);
    else { s = s * __expf(m - u) + 1.f; m = u; }
    hb[idx] = f2bf(__expf(a + m + __logf(s)));   // h[t]
    idx += DIM_INNER;
  }
}

// ---- Kernel 5: GEMM2  out = h @ W_out  -----------------------------------
// Block = 8 waves sharing one 64-col N tile; B tile staged to LDS with
// double-buffered global_load_async_to_lds_b128 (ASYNCcnt path).
__global__ __launch_bounds__(256) void mg_gemm2(const uint16_t* __restrict__ hb,     // [M_TOT, DIM_INNER]
                                                const uint16_t* __restrict__ woutT,  // [DIMX, DIM_INNER]
                                                float* __restrict__ out) {           // [M_TOT, DIMX]
  __shared__ alignas(16) uint16_t btile[2][64 * 32];   // 2 x 4KB double buffer
  const int tid  = threadIdx.x;
  const int lane = tid & 31;
  const int wav  = tid >> 5;
  const int nTile = blockIdx.x & 15;                   // DIMX/64 = 16 n tiles
  const int m0 = (((blockIdx.x >> 4) << 3) + wav) * 64;
  const int n0 = nTile * 64;

  // Each thread async-copies 16B of the 64x32 B tile: row = tid/4, chunk = tid%4
  const int brow = tid >> 2;
  const int bchk = tid & 3;
  const uint32_t lds0 = (uint32_t)(uintptr_t)(&btile[0][0]);
  const uint32_t lds1 = (uint32_t)(uintptr_t)(&btile[1][0]);
  const uint32_t ldsLocal = (uint32_t)(brow * 64 + bchk * 16);  // bytes within buffer

  auto issue = [&](int buf, int k0) {
    const uint64_t ga = (uint64_t)(uintptr_t)(woutT + (size_t)(n0 + brow) * DIM_INNER + k0 + bchk * 8);
    const uint32_t la = (buf ? lds1 : lds0) + ldsLocal;
    asm volatile("global_load_async_to_lds_b128 %0, %1, off"
                 :: "v"(la), "v"(ga) : "memory");
  };

  const v8f vzero = {0.f, 0.f, 0.f, 0.f, 0.f, 0.f, 0.f, 0.f};
  v8f acc[4][4];
  #pragma unroll
  for (int mi = 0; mi < 4; ++mi)
    #pragma unroll
    for (int ni = 0; ni < 4; ++ni) acc[mi][ni] = vzero;

  issue(0, 0);
  const int NK = DIM_INNER / 32;   // 64 k-steps
  for (int ki = 0; ki < NK; ++ki) {
    const int k0 = ki * 32;
    if (ki + 1 < NK) {
      issue((ki + 1) & 1, k0 + 32);
      asm volatile("s_wait_asynccnt 0x1" ::: "memory");  // current buffer ready
    } else {
      asm volatile("s_wait_asynccnt 0x0" ::: "memory");
    }
    __syncthreads();

    __builtin_prefetch(hb + (size_t)(m0 + (lane & 15)) * DIM_INNER + k0 + 32, 0, 0);
    v16bf a[4];
    #pragma unroll
    for (int mi = 0; mi < 4; ++mi)
      a[mi] = load_frag_a(hb, DIM_INNER, m0 + 16 * mi, k0, lane);

    const uint16_t* bt = &btile[ki & 1][0];   // LDS, row-major [64][32]
    #pragma unroll
    for (int ni = 0; ni < 4; ++ni) {
      const int n  = lane & 15;
      const int kh = lane >> 4;
      const uint16_t* p = bt + (ni * 16 + n) * 32 + kh * 16;
      FragRaw f;
      f.lo = *reinterpret_cast<const uint4*>(p);
      f.hi = *reinterpret_cast<const uint4*>(p + 8);
      const v16bf bfrag = __builtin_bit_cast(v16bf, f);
      #pragma unroll
      for (int mi = 0; mi < 4; ++mi)
        acc[mi][ni] = __builtin_amdgcn_wmma_f32_16x16x32_bf16(
            false, a[mi], false, bfrag, (short)0, acc[mi][ni], false, false);
    }
    __syncthreads();   // reads done before next overwrite of this buffer
  }

  const int cn = lane & 15;
  const int cm = (lane >> 4) * 8;
  #pragma unroll
  for (int mi = 0; mi < 4; ++mi)
    #pragma unroll
    for (int ni = 0; ni < 4; ++ni)
      #pragma unroll
      for (int i = 0; i < 8; ++i)
        out[(size_t)(m0 + 16 * mi + cm + i) * DIMX + (n0 + 16 * ni + cn)] = acc[mi][ni][i];
}

// ---- Host-side launcher ---------------------------------------------------
extern "C" void kernel_launch(void* const* d_in, const int* in_sizes, int n_in,
                              void* d_out, int out_size, void* d_ws, size_t ws_size,
                              hipStream_t stream) {
  (void)in_sizes; (void)n_in; (void)out_size; (void)ws_size;
  const float* x    = (const float*)d_in[0];   // [4,4096,1024]
  const float* Whg  = (const float*)d_in[1];   // [1024,4096]
  const float* Wout = (const float*)d_in[2];   // [2048,1024]
  float*       out  = (float*)d_out;           // [4,4096,1024]

  // Workspace layout (~366 MB total)
  char* ws = (char*)d_ws;
  const size_t MB = 1024ull * 1024ull;
  uint16_t* xb     = (uint16_t*)(ws);                   //  32 MB: x bf16 [16384,1024]
  uint16_t* whgT   = (uint16_t*)(ws +  32 * MB);        //   8 MB: W_hg^T bf16 [4096,1024]
  uint16_t* woutT  = (uint16_t*)(ws +  40 * MB);        //   4 MB: W_out^T bf16 [1024,2048]
  float*    logc   = (float*)   (ws +  44 * MB);        // 128 MB: [16384,2048]
  float*    logv   = (float*)   (ws + 172 * MB);        // 128 MB: [16384,2048]
  uint16_t* hb     = (uint16_t*)(ws + 300 * MB);        //  64 MB: h bf16 [16384,2048]
  float*    chunkA = (float*)   (ws + 364 * MB);        // 512 KB
  float*    chunkL = (float*)   (ws + 364 * MB + 512 * 1024);
  float*    carryA = (float*)   (ws + 365 * MB);        // 512 KB
  float*    carryG = (float*)   (ws + 365 * MB + 512 * 1024);

  // 1) x -> bf16
  mg_convert_x<<<16384, 256, 0, stream>>>(x, xb);
  // 2) weight transpose+convert
  mg_transpose_bf16<<<dim3(NHG / 32, DIMX / 32), dim3(32, 8), 0, stream>>>(Whg, whgT, DIMX, NHG);
  mg_transpose_bf16<<<dim3(DIMX / 32, DIM_INNER / 32), dim3(32, 8), 0, stream>>>(Wout, woutT, DIM_INNER, DIMX);
  // 3) GEMM1 + activations: (16384/64)*(2048/32) = 16384 waves / 8 per block
  mg_gemm1_act<<<2048, 256, 0, stream>>>(xb, whgT, logc, logv);
  // 4) chunked scan: 8192 channels x 16 chunks
  mg_scan_part <<<(NCH * NCHUNK) / 256, 256, 0, stream>>>(logc, logv, chunkA, chunkL);
  mg_scan_carry<<<NCH / 256, 256, 0, stream>>>(chunkA, chunkL, carryA, carryG);
  mg_scan_apply<<<(NCH * NCHUNK) / 256, 256, 0, stream>>>(logc, logv, carryA, carryG, hb);
  // 5) GEMM2 with async-to-LDS double-buffered B tiles
  mg_gemm2<<<512, 256, 0, stream>>>(hb, woutT, out);
}
